// CharLevelLanguageModel_90254442758697
// MI455X (gfx1250) — compile-verified, matrix operand order checked by reading, not run on hardware
//
#include <hip/hip_runtime.h>
#include <hip/hip_bf16.h>

typedef __attribute__((ext_vector_type(16))) _Float16 v16h;
typedef __attribute__((ext_vector_type(8)))  _Float16 v8h;
typedef __attribute__((ext_vector_type(8)))  float    v8f;

// Model dims
#define NTOK 16384   // B*T
#define BB   64
#define TT   256
#define CD   384
#define NH   6
#define NL   6
#define NV   65
#define NVP  128     // padded LM-head rows (zero-filled) so B loads need no guard
#define HS   64
#define DF   1536
#define LDQ  1152    // 3*C, qkv packed columns

// ---------------------------------------------------------------------------
// WMMA helpers (gfx1250, wave32)
// ---------------------------------------------------------------------------
__device__ __forceinline__ v8f wmma_f16(v16h a, v16h b, v8f c) {
  return __builtin_amdgcn_wmma_f32_16x16x32_f16(false, a, false, b, (short)0, c,
                                                false, false);
}

__device__ __forceinline__ v8f zero8() {
  v8f z;
#pragma unroll
  for (int i = 0; i < 8; ++i) z[i] = 0.f;
  return z;
}

// Load a 16x32 f16 fragment (A layout; also B layout when source is [n,k]
// row-major).  base points at element (row 0, k 0); ld in halfs (mult of 8).
__device__ __forceinline__ v16h load_frag16(const _Float16* base, int ld) {
  int lane = threadIdx.x & 31;
  const _Float16* p = base + (size_t)(lane & 15) * ld + ((lane >> 4) << 3);
  v16h r;
  *(v8h*)&r       = *(const v8h*)p;         // K 0..7   (or 8..15 upper half)
  *((v8h*)&r + 1) = *(const v8h*)(p + 16);  // K 16..23 (or 24..31)
  return r;
}

__device__ __forceinline__ float redmax16(float x) {
  x = fmaxf(x, __shfl_xor(x, 1, 16));
  x = fmaxf(x, __shfl_xor(x, 2, 16));
  x = fmaxf(x, __shfl_xor(x, 4, 16));
  x = fmaxf(x, __shfl_xor(x, 8, 16));
  return x;
}
__device__ __forceinline__ float redsum16(float x) {
  x += __shfl_xor(x, 1, 16);
  x += __shfl_xor(x, 2, 16);
  x += __shfl_xor(x, 4, 16);
  x += __shfl_xor(x, 8, 16);
  return x;
}

// ---------------------------------------------------------------------------
// Weight packing: f32 -> f16, transposed to [out_col, in_k] row-major
// ---------------------------------------------------------------------------
__global__ void pack_qkv_kernel(const float* __restrict__ Wq,
                                const float* __restrict__ Wk,
                                const float* __restrict__ Wv,
                                _Float16* __restrict__ out) {
  size_t i = (size_t)blockIdx.x * blockDim.x + threadIdx.x;
  const size_t total = (size_t)NL * LDQ * CD;
  if (i >= total) return;
  int c  = (int)(i % CD);
  int j  = (int)((i / CD) % LDQ);
  int l  = (int)(i / ((size_t)CD * LDQ));
  int which = j / CD;            // 0=q 1=k 2=v
  int jj = j % CD;
  int h = jj / HS, d = jj % HS;
  const float* W = (which == 0) ? Wq : (which == 1) ? Wk : Wv;
  out[i] = (_Float16)W[(((size_t)l * NH + h) * CD + c) * HS + d];
}

// in: [L, K, Nc_src] row-major -> out: [L, Nc_out, K] f16 (rows >= Nc_src = 0)
__global__ void pack_t_kernel(const float* __restrict__ in,
                              _Float16* __restrict__ out, int K, int Nc_src,
                              int Nc_out, size_t total) {
  size_t i = (size_t)blockIdx.x * blockDim.x + threadIdx.x;
  if (i >= total) return;
  int k = (int)(i % K);
  int j = (int)((i / K) % Nc_out);
  int l = (int)(i / ((size_t)K * Nc_out));
  out[i] = (j < Nc_src) ? (_Float16)in[((size_t)l * K + k) * Nc_src + j]
                        : (_Float16)0.f;
}

// ---------------------------------------------------------------------------
// Embedding: x[n,c] = tok_emb[idx[n],c] + pos_emb[n%T, c]
// ---------------------------------------------------------------------------
__global__ void embed_kernel(const int* __restrict__ idx,
                             const float* __restrict__ tok,
                             const float* __restrict__ pos,
                             float* __restrict__ x) {
  size_t i = (size_t)blockIdx.x * blockDim.x + threadIdx.x;
  int c = (int)(i % CD);
  int n = (int)(i / CD);
  int t = n % TT;
  x[i] = tok[(size_t)idx[n] * CD + c] + pos[(size_t)t * CD + c];
}

// ---------------------------------------------------------------------------
// LayerNorm (one wave32 per row of C=384), f32 in -> f16 out
// ---------------------------------------------------------------------------
__global__ __launch_bounds__(256) void ln_kernel(const float* __restrict__ x,
                                                 const float* __restrict__ g,
                                                 const float* __restrict__ b,
                                                 _Float16* __restrict__ out) {
  int row  = blockIdx.x * 8 + (threadIdx.x >> 5);
  int lane = threadIdx.x & 31;
  const float* r = x + (size_t)row * CD;
  float s = 0.f;
  for (int c = lane; c < CD; c += 32) s += r[c];
#pragma unroll
  for (int m = 16; m >= 1; m >>= 1) s += __shfl_xor(s, m, 32);
  float mu = s * (1.f / CD);
  float v = 0.f;
  for (int c = lane; c < CD; c += 32) { float d = r[c] - mu; v += d * d; }
#pragma unroll
  for (int m = 16; m >= 1; m >>= 1) v += __shfl_xor(v, m, 32);
  float rstd = rsqrtf(v * (1.f / CD) + 1e-5f);
  _Float16* o = out + (size_t)row * CD;
  for (int c = lane; c < CD; c += 32)
    o[c] = (_Float16)((r[c] - mu) * rstd * g[c] + b[c]);
}

// ---------------------------------------------------------------------------
// Generic WMMA GEMM: out[M,Ncols] = A[M,K] @ Bt[*,K]^T (+bias)(relu)(+res)
// 8 waves/block, wave tile 32x64, block tile 128x128, 2-stage K pipeline.
// Bt must have >= roundup(Ncols,128) valid (zero-padded) rows.
// ---------------------------------------------------------------------------
__global__ __launch_bounds__(256) void gemm_kernel(
    const _Float16* __restrict__ A, int lda, const _Float16* __restrict__ Bt,
    int ldb, const float* __restrict__ bias, const float* __restrict__ res,
    float* __restrict__ outF, _Float16* __restrict__ outH, int ldo, int Ncols,
    int K, int relu) {
  int lane = threadIdx.x & 31;
  int wv = threadIdx.x >> 5;
  int m0 = blockIdx.y * 128 + (wv >> 1) * 32;
  int n0 = blockIdx.x * 128 + (wv & 1) * 64;

  const _Float16* Ap0 = A + (size_t)m0 * lda;
  const _Float16* Ap1 = A + (size_t)(m0 + 16) * lda;
  const _Float16* Bp0 = Bt + (size_t)n0 * ldb;
  const _Float16* Bp1 = Bt + (size_t)(n0 + 16) * ldb;
  const _Float16* Bp2 = Bt + (size_t)(n0 + 32) * ldb;
  const _Float16* Bp3 = Bt + (size_t)(n0 + 48) * ldb;

  v8f acc[2][4];
#pragma unroll
  for (int i = 0; i < 2; ++i)
#pragma unroll
    for (int j = 0; j < 4; ++j) acc[i][j] = zero8();

  // prologue loads (k = 0)
  v16h a0 = load_frag16(Ap0, lda);
  v16h a1 = load_frag16(Ap1, lda);
  v16h b0 = load_frag16(Bp0, ldb);
  v16h b1 = load_frag16(Bp1, ldb);
  v16h b2 = load_frag16(Bp2, ldb);
  v16h b3 = load_frag16(Bp3, ldb);

  for (int k = 32; k < K; k += 32) {
    // prefetch next k-block before consuming current one
    v16h na0 = load_frag16(Ap0 + k, lda);
    v16h na1 = load_frag16(Ap1 + k, lda);
    v16h nb0 = load_frag16(Bp0 + k, ldb);
    v16h nb1 = load_frag16(Bp1 + k, ldb);
    v16h nb2 = load_frag16(Bp2 + k, ldb);
    v16h nb3 = load_frag16(Bp3 + k, ldb);
    acc[0][0] = wmma_f16(a0, b0, acc[0][0]);
    acc[0][1] = wmma_f16(a0, b1, acc[0][1]);
    acc[0][2] = wmma_f16(a0, b2, acc[0][2]);
    acc[0][3] = wmma_f16(a0, b3, acc[0][3]);
    acc[1][0] = wmma_f16(a1, b0, acc[1][0]);
    acc[1][1] = wmma_f16(a1, b1, acc[1][1]);
    acc[1][2] = wmma_f16(a1, b2, acc[1][2]);
    acc[1][3] = wmma_f16(a1, b3, acc[1][3]);
    a0 = na0; a1 = na1;
    b0 = nb0; b1 = nb1; b2 = nb2; b3 = nb3;
  }
  // epilogue of pipeline
  acc[0][0] = wmma_f16(a0, b0, acc[0][0]);
  acc[0][1] = wmma_f16(a0, b1, acc[0][1]);
  acc[0][2] = wmma_f16(a0, b2, acc[0][2]);
  acc[0][3] = wmma_f16(a0, b3, acc[0][3]);
  acc[1][0] = wmma_f16(a1, b0, acc[1][0]);
  acc[1][1] = wmma_f16(a1, b1, acc[1][1]);
  acc[1][2] = wmma_f16(a1, b2, acc[1][2]);
  acc[1][3] = wmma_f16(a1, b3, acc[1][3]);

  int laneHi = lane >> 4, nrel = lane & 15;
#pragma unroll
  for (int i = 0; i < 2; ++i)
#pragma unroll
    for (int j = 0; j < 4; ++j) {
      int col = n0 + j * 16 + nrel;
      if (col >= Ncols) continue;
      float bb = bias ? bias[col] : 0.f;
#pragma unroll
      for (int e = 0; e < 8; ++e) {
        int m = m0 + i * 16 + e + 8 * laneHi;
        float v = acc[i][j][e] + bb;
        if (relu) v = fmaxf(v, 0.f);
        if (res) v += res[(size_t)m * ldo + col];
        if (outF) outF[(size_t)m * ldo + col] = v;
        if (outH) outH[(size_t)m * ldo + col] = (_Float16)v;
      }
    }
}

// ---------------------------------------------------------------------------
// V transpose: qkv [N, 3C] -> vT [(b,h,d), s]
// ---------------------------------------------------------------------------
__global__ void transpose_v_kernel(const _Float16* __restrict__ qkv,
                                   _Float16* __restrict__ vT) {
  size_t i = (size_t)blockIdx.x * blockDim.x + threadIdx.x;
  int s = (int)(i % TT);
  int d = (int)((i / TT) % HS);
  int h = (int)((i / ((size_t)TT * HS)) % NH);
  int b = (int)(i / ((size_t)TT * HS * NH));
  vT[i] = qkv[(size_t)(b * TT + s) * LDQ + 2 * CD + h * HS + d];
}

// ---------------------------------------------------------------------------
// Flash attention: one wave per (b, h, 16-row q tile); online softmax over
// 32-column K/V blocks; P is bounced through LDS to get A-fragment layout.
// ---------------------------------------------------------------------------
__global__ __launch_bounds__(256) void attn_kernel(
    const _Float16* __restrict__ qkv, const _Float16* __restrict__ vT,
    _Float16* __restrict__ att) {
  __shared__ _Float16 sh[8][16 * 40];
  int lane = threadIdx.x & 31;
  int wv = threadIdx.x >> 5;
  int wid = blockIdx.x * 8 + wv;
  int tt = wid & 15;
  int h = (wid >> 4) % NH;
  int b = wid / (16 * NH);
  int t0 = tt * 16;
  int laneHi = lane >> 4, nrel = lane & 15;

  const _Float16* qbase = qkv + (size_t)(b * TT + t0) * LDQ + h * HS;
  const _Float16* kbase = qkv + (size_t)(b * TT) * LDQ + CD + h * HS;
  const _Float16* vtb = vT + (size_t)(b * NH + h) * HS * TT;

  v16h qf0 = load_frag16(qbase, LDQ);        // k 0..31
  v16h qf1 = load_frag16(qbase + 32, LDQ);   // k 32..63

  v8f o0 = zero8(), o1 = zero8(), o2 = zero8(), o3 = zero8();
  float Mrow[8], Lrow[8];
#pragma unroll
  for (int e = 0; e < 8; ++e) { Mrow[e] = -1e30f; Lrow[e] = 0.f; }

  for (int s0 = 0; s0 < t0 + 16; s0 += 32) {
    // S = (Q K^T) * scale over 32 key rows -> two 16x16 f32 frags
    v8f sc0 = zero8(), sc1 = zero8();
    v16h kb;
    kb = load_frag16(kbase + (size_t)s0 * LDQ, LDQ);
    sc0 = wmma_f16(qf0, kb, sc0);
    kb = load_frag16(kbase + (size_t)s0 * LDQ + 32, LDQ);
    sc0 = wmma_f16(qf1, kb, sc0);
    kb = load_frag16(kbase + (size_t)(s0 + 16) * LDQ, LDQ);
    sc1 = wmma_f16(qf0, kb, sc1);
    kb = load_frag16(kbase + (size_t)(s0 + 16) * LDQ + 32, LDQ);
    sc1 = wmma_f16(qf1, kb, sc1);

    const float scale = 0.125f;  // HS^-0.5
#pragma unroll
    for (int e = 0; e < 8; ++e) { sc0[e] *= scale; sc1[e] *= scale; }

    if (s0 + 31 > t0) {  // causal mask needed
#pragma unroll
      for (int e = 0; e < 8; ++e) {
        int m = t0 + e + 8 * laneHi;
        if (s0 + nrel > m) sc0[e] = -1e30f;
        if (s0 + 16 + nrel > m) sc1[e] = -1e30f;
      }
    }

    // online softmax row update
    float alpha[8];
#pragma unroll
    for (int e = 0; e < 8; ++e) {
      float mx = redmax16(fmaxf(sc0[e], sc1[e]));
      float Mn = fmaxf(Mrow[e], mx);
      alpha[e] = __expf(Mrow[e] - Mn);
      Mrow[e] = Mn;
      float p0 = __expf(sc0[e] - Mn);
      float p1 = __expf(sc1[e] - Mn);
      sc0[e] = p0; sc1[e] = p1;
      Lrow[e] = Lrow[e] * alpha[e] + redsum16(p0 + p1);
    }
#pragma unroll
    for (int e = 0; e < 8; ++e) {
      o0[e] *= alpha[e]; o1[e] *= alpha[e];
      o2[e] *= alpha[e]; o3[e] *= alpha[e];
    }

    // C/D layout -> A layout via LDS (f16, padded stride 40 halfs)
    _Float16* myl = &sh[wv][0];
#pragma unroll
    for (int e = 0; e < 8; ++e) {
      int m = e + 8 * laneHi;
      myl[m * 40 + nrel] = (_Float16)sc0[e];
      myl[m * 40 + 16 + nrel] = (_Float16)sc1[e];
    }
    v16h pa;
    {
      const _Float16* p = myl + (lane & 15) * 40 + ((lane >> 4) << 3);
#pragma unroll
      for (int e = 0; e < 8; ++e) { pa[e] = p[e]; pa[8 + e] = p[16 + e]; }
    }

    // O += P @ V  (V^T rows are output dims, contiguous in s)
    o0 = wmma_f16(pa, load_frag16(vtb + (size_t)(0 * 16) * TT + s0, TT), o0);
    o1 = wmma_f16(pa, load_frag16(vtb + (size_t)(1 * 16) * TT + s0, TT), o1);
    o2 = wmma_f16(pa, load_frag16(vtb + (size_t)(2 * 16) * TT + s0, TT), o2);
    o3 = wmma_f16(pa, load_frag16(vtb + (size_t)(3 * 16) * TT + s0, TT), o3);
  }

  _Float16* obase = att + (size_t)(b * TT + t0) * CD + h * HS;
#pragma unroll
  for (int e = 0; e < 8; ++e) {
    int m = e + 8 * laneHi;
    float inv = 1.f / Lrow[e];
    obase[(size_t)m * CD + 0 * 16 + nrel] = (_Float16)(o0[e] * inv);
    obase[(size_t)m * CD + 1 * 16 + nrel] = (_Float16)(o1[e] * inv);
    obase[(size_t)m * CD + 2 * 16 + nrel] = (_Float16)(o2[e] * inv);
    obase[(size_t)m * CD + 3 * 16 + nrel] = (_Float16)(o3[e] * inv);
  }
}

// ---------------------------------------------------------------------------
// Host launcher
// ---------------------------------------------------------------------------
extern "C" void kernel_launch(void* const* d_in, const int* in_sizes, int n_in,
                              void* d_out, int out_size, void* d_ws,
                              size_t ws_size, hipStream_t stream) {
  (void)in_sizes; (void)n_in; (void)out_size; (void)ws_size;
  const int*   idx   = (const int*)d_in[0];
  const float* tok   = (const float*)d_in[1];
  const float* pos   = (const float*)d_in[2];
  const float* Wq    = (const float*)d_in[3];
  const float* Wk    = (const float*)d_in[4];
  const float* Wv    = (const float*)d_in[5];
  const float* Wo    = (const float*)d_in[6];
  const float* bo    = (const float*)d_in[7];
  const float* ln1g  = (const float*)d_in[8];
  const float* ln1b  = (const float*)d_in[9];
  const float* ln2g  = (const float*)d_in[10];
  const float* ln2b  = (const float*)d_in[11];
  const float* W1    = (const float*)d_in[12];
  const float* b1    = (const float*)d_in[13];
  const float* W2    = (const float*)d_in[14];
  const float* b2    = (const float*)d_in[15];
  const float* lnfg  = (const float*)d_in[16];
  const float* lnfb  = (const float*)d_in[17];
  const float* Wlm   = (const float*)d_in[18];
  const float* blm   = (const float*)d_in[19];

  char* ws = (char*)d_ws;
  size_t off = 0;
  auto alloc = [&](size_t bytes) -> char* {
    char* p = ws + off;
    off = (off + bytes + 255) & ~(size_t)255;
    return p;
  };

  float*    x     = (float*)alloc((size_t)NTOK * CD * 4);
  _Float16* h16   = (_Float16*)alloc((size_t)NTOK * CD * 2);
  _Float16* qkv16 = (_Float16*)alloc((size_t)NTOK * LDQ * 2);
  _Float16* vT    = (_Float16*)alloc((size_t)NTOK * CD * 2);
  _Float16* att16 = (_Float16*)alloc((size_t)NTOK * CD * 2);
  _Float16* mlp16 = (_Float16*)alloc((size_t)NTOK * DF * 2);
  _Float16* WqkvT = (_Float16*)alloc((size_t)NL * LDQ * CD * 2);
  _Float16* WoT   = (_Float16*)alloc((size_t)NL * CD * CD * 2);
  _Float16* W1T   = (_Float16*)alloc((size_t)NL * DF * CD * 2);
  _Float16* W2T   = (_Float16*)alloc((size_t)NL * CD * DF * 2);
  _Float16* WlmT  = (_Float16*)alloc((size_t)NVP * CD * 2);  // zero-padded rows

  // Pack all weights to f16 transposed layouts.
  {
    size_t t = (size_t)NL * LDQ * CD;
    pack_qkv_kernel<<<(unsigned)((t + 255) / 256), 256, 0, stream>>>(Wq, Wk, Wv,
                                                                     WqkvT);
  }
  {
    size_t t = (size_t)NL * CD * CD;
    pack_t_kernel<<<(unsigned)((t + 255) / 256), 256, 0, stream>>>(Wo, WoT, CD,
                                                                   CD, CD, t);
  }
  {
    size_t t = (size_t)NL * DF * CD;  // W1: [L,C,DFF] -> [L,DFF,C]
    pack_t_kernel<<<(unsigned)((t + 255) / 256), 256, 0, stream>>>(W1, W1T, CD,
                                                                   DF, DF, t);
  }
  {
    size_t t = (size_t)NL * CD * DF;  // W2: [L,DFF,C] -> [L,C,DFF]
    pack_t_kernel<<<(unsigned)((t + 255) / 256), 256, 0, stream>>>(W2, W2T, DF,
                                                                   CD, CD, t);
  }
  {
    size_t t = (size_t)NVP * CD;  // Wlm: [C,V] -> [128,C] zero-padded
    pack_t_kernel<<<(unsigned)((t + 255) / 256), 256, 0, stream>>>(Wlm, WlmT,
                                                                   CD, NV, NVP,
                                                                   t);
  }

  embed_kernel<<<NTOK * CD / 256, 256, 0, stream>>>(idx, tok, pos, x);

  for (int l = 0; l < NL; ++l) {
    ln_kernel<<<NTOK / 8, 256, 0, stream>>>(x, ln1g + l * CD, ln1b + l * CD,
                                            h16);
    {  // QKV projection -> qkv16 [N, 3C] f16
      dim3 g(LDQ / 128, NTOK / 128);
      gemm_kernel<<<g, 256, 0, stream>>>(h16, CD,
                                         WqkvT + (size_t)l * LDQ * CD, CD,
                                         nullptr, nullptr, nullptr, qkv16, LDQ,
                                         LDQ, CD, 0);
    }
    transpose_v_kernel<<<NTOK * CD / 256, 256, 0, stream>>>(qkv16, vT);
    attn_kernel<<<(BB * NH * (TT / 16)) / 8, 256, 0, stream>>>(qkv16, vT,
                                                               att16);
    {  // out projection + bias + residual -> x (f32, in place)
      dim3 g(CD / 128, NTOK / 128);
      gemm_kernel<<<g, 256, 0, stream>>>(att16, CD,
                                         WoT + (size_t)l * CD * CD, CD,
                                         bo + l * CD, x, x, nullptr, CD, CD,
                                         CD, 0);
    }
    ln_kernel<<<NTOK / 8, 256, 0, stream>>>(x, ln2g + l * CD, ln2b + l * CD,
                                            h16);
    {  // MLP up + ReLU -> mlp16 f16
      dim3 g(DF / 128, NTOK / 128);
      gemm_kernel<<<g, 256, 0, stream>>>(h16, CD, W1T + (size_t)l * DF * CD,
                                         CD, b1 + l * DF, nullptr, nullptr,
                                         mlp16, DF, DF, CD, 1);
    }
    {  // MLP down + bias + residual -> x
      dim3 g(CD / 128, NTOK / 128);
      gemm_kernel<<<g, 256, 0, stream>>>(mlp16, DF, W2T + (size_t)l * CD * DF,
                                         DF, b2 + l * CD, x, x, nullptr, CD,
                                         CD, DF, 0);
    }
  }

  ln_kernel<<<NTOK / 8, 256, 0, stream>>>(x, lnfg, lnfb, h16);
  {  // LM head -> logits f32 [N, V]; WlmT rows padded to 128 so loads are safe
    dim3 g(1, NTOK / 128);
    gemm_kernel<<<g, 256, 0, stream>>>(h16, CD, WlmT, CD, blm, nullptr,
                                       (float*)d_out, nullptr, NV, NV, CD, 0);
  }
}
